// HGTNet_4879082848506
// MI455X (gfx1250) — matrix operand order repaired and checked by reading, not compiled, not run on hardware
//
#include <hip/hip_runtime.h>
#include <hip/hip_bf16.h>
#include <math.h>

// ---------------------------------------------------------------------------
// Problem constants (match reference)
// ---------------------------------------------------------------------------
#define NA_   100000   // multiple of 16 -> no row guards in GEMM
#define NP__  100000
#define DD    128
#define HH    4
#define DHH   32
#define LL    2
#define EE    400000
#define OUTD  40

typedef __attribute__((ext_vector_type(16))) _Float16 v16h;
typedef __attribute__((ext_vector_type(8)))  _Float16 v8h;
typedef __attribute__((ext_vector_type(8)))  float    v8f;

__device__ __forceinline__ float gelu_exact(float x)
{
    return 0.5f * x * (1.0f + erff(x * 0.70710678118654752f));
}

// Stage 16x128 f32 tile -> f16 LDS (optionally exact-gelu), vectorized:
// 256 threads x 8 elements, 2x float4 global loads + 1x b128 LDS store each.
template<bool GELU_A>
__device__ __forceinline__ void stage_A(const float* __restrict__ X, int row0,
                                        _Float16 (*As)[128], int tid)
{
    const int r = tid >> 4;            // 0..15
    const int c = (tid & 15) * 8;      // 0..120 step 8
    const float4 x0 = *(const float4*)&X[(size_t)(row0 + r) * DD + c];
    const float4 x1 = *(const float4*)&X[(size_t)(row0 + r) * DD + c + 4];
    float v[8] = {x0.x, x0.y, x0.z, x0.w, x1.x, x1.y, x1.z, x1.w};
    v8h hv;
#pragma unroll
    for (int j = 0; j < 8; ++j) {
        float f = v[j];
        if (GELU_A) f = gelu_exact(f);
        hv[j] = (_Float16)f;
    }
    *(v8h*)&As[r][c] = hv;
}

// Stage this wave's 128x16 B slice of W[128,NC] transposed into Bs[col][k].
template<int NC>
__device__ __forceinline__ void stage_B(const float* __restrict__ W, int col0,
                                        _Float16 (*Bs)[128], int lane)
{
#pragma unroll
    for (int it = 0; it < 16; ++it) {
        const int chunk = lane + it * 32;
        const int k  = chunk >> 2;
        const int c4 = (chunk & 3) * 4;
        float4 wv = {0.f, 0.f, 0.f, 0.f};
        if (NC % 128 == 0 || col0 + c4 + 3 < NC)
            wv = *(const float4*)&W[(size_t)k * NC + col0 + c4];
        Bs[c4 + 0][k] = (_Float16)wv.x;
        Bs[c4 + 1][k] = (_Float16)wv.y;
        Bs[c4 + 2][k] = (_Float16)wv.z;
        Bs[c4 + 3][k] = (_Float16)wv.w;
    }
}

__device__ __forceinline__ v16h load_Afrag(const _Float16 (*As)[128],
                                           int rsel, int half, int kb)
{
    v8h alo = *(const v8h*)&As[rsel][kb + half * 16];
    v8h ahi = *(const v8h*)&As[rsel][kb + half * 16 + 8];
    v16h a;
#pragma unroll
    for (int i = 0; i < 8; ++i) { a[i] = alo[i]; a[8 + i] = ahi[i]; }
    return a;
}

// ---------------------------------------------------------------------------
// WMMA GEMM:  Y[M,NC] = act(X)[M,128] @ W[128,NC] (+bias via acc init)
// SKIP_ELU: Y holds h (in/out): h = elu(s*o + (1-s)*h), s = sigmoid(skip[0])
// ---------------------------------------------------------------------------
template<int NC, bool GELU_A, bool SKIP_ELU>
__global__ __launch_bounds__(256)
void gemm128(const float* __restrict__ X, const float* __restrict__ W,
             const float* __restrict__ bias, float* __restrict__ Y,
             const float* __restrict__ skip)
{
    __shared__ _Float16 As[16][128];      // 4 KB
    __shared__ _Float16 Bs[8][16][128];   // 32 KB [wave][col][k]

    const int tid  = threadIdx.x;
    const int wave = tid >> 5;
    const int lane = tid & 31;
    const int row0 = blockIdx.x * 16;
    const int col0 = wave * 16;

    stage_A<GELU_A>(X, row0, As, tid);
    __syncthreads();

    if (col0 < NC) {                       // wave-uniform; EXEC stays full
        stage_B<NC>(W, col0, Bs[wave], lane);

        const int half = lane >> 4;
        const int rsel = lane & 15;
        const int col  = col0 + rsel;

        const float bv = (NC % 16 == 0 || col < NC) ? bias[col] : 0.0f;
        v8f acc;
#pragma unroll
        for (int i = 0; i < 8; ++i) acc[i] = bv;

#pragma unroll
        for (int ks = 0; ks < 4; ++ks) {
            const int kb = ks * 32;
            v16h a = load_Afrag(As, rsel, half, kb);
            v16h b = *(const v16h*)&Bs[wave][rsel][kb + half * 16];
            acc = __builtin_amdgcn_wmma_f32_16x16x32_f16(
                      false, a, false, b, (short)0, acc, false, false);
        }

        if (SKIP_ELU) {
            const float s = 1.0f / (1.0f + __expf(-skip[0]));
#pragma unroll
            for (int i = 0; i < 8; ++i) {
                const int r = row0 + i + half * 8;
                float* yp = &Y[(size_t)r * NC + col];
                const float v = s * acc[i] + (1.0f - s) * (*yp);
                *yp = v > 0.0f ? v : (__expf(v) - 1.0f);
            }
        } else {
#pragma unroll
            for (int i = 0; i < 8; ++i) {
                const int r = row0 + i + half * 8;
                if (NC % 16 == 0 || col < NC)
                    Y[(size_t)r * NC + col] = acc[i];
            }
        }
    }
}

// ---------------------------------------------------------------------------
// Fused K/Q/V: stage activations once, apply 3 weight matrices.
// ---------------------------------------------------------------------------
__global__ __launch_bounds__(256)
void kqv_wmma(const float* __restrict__ X, const float* __restrict__ Wbase,
              const float* __restrict__ Bbase, float* __restrict__ K,
              float* __restrict__ Q, float* __restrict__ V)
{
    __shared__ _Float16 As[16][128];
    __shared__ _Float16 Bs[8][16][128];

    const int tid  = threadIdx.x;
    const int wave = tid >> 5;
    const int lane = tid & 31;
    const int row0 = blockIdx.x * 16;
    const int col0 = wave * 16;
    const int half = lane >> 4;
    const int rsel = lane & 15;
    const int col  = col0 + rsel;

    stage_A<false>(X, row0, As, tid);
    __syncthreads();

    v16h af[4];
#pragma unroll
    for (int ks = 0; ks < 4; ++ks) af[ks] = load_Afrag(As, rsel, half, ks * 32);

    float* const outs[3] = {K, Q, V};
#pragma unroll
    for (int i = 0; i < 3; ++i) {
        stage_B<DD>(Wbase + (size_t)i * 2 * DD * DD, col0, Bs[wave], lane);

        const float bv = Bbase[(size_t)i * 2 * DD + col];
        v8f acc;
#pragma unroll
        for (int j = 0; j < 8; ++j) acc[j] = bv;

#pragma unroll
        for (int ks = 0; ks < 4; ++ks) {
            v16h b = *(const v16h*)&Bs[wave][rsel][ks * 32 + half * 16];
            acc = __builtin_amdgcn_wmma_f32_16x16x32_f16(
                      false, af[ks], false, b, (short)0, acc, false, false);
        }
#pragma unroll
        for (int j = 0; j < 8; ++j)
            outs[i][(size_t)(row0 + j + half * 8) * DD + col] = acc[j];
    }
}

// ---------------------------------------------------------------------------
// Per-head block-diagonal projection (hoists edge transforms to nodes):
//   Y[n, h*32+f] = sum_d X[n, h*32+d] * R[h, d, f]
// wave -> (head = wave/2, f-halftile = (wave&1)*16) ; single K=32 WMMA.
// ---------------------------------------------------------------------------
__global__ __launch_bounds__(256)
void headproj_wmma(const float* __restrict__ X, const float* __restrict__ R,
                   float* __restrict__ Y)
{
    __shared__ _Float16 As[16][128];      // 4 KB
    __shared__ _Float16 Bs[8][16][32];    // 8 KB [wave][f][d]

    const int tid  = threadIdx.x;
    const int wave = tid >> 5;
    const int lane = tid & 31;
    const int row0 = blockIdx.x * 16;
    const int head = wave >> 1;
    const int fc   = (wave & 1) * 16;

    stage_A<false>(X, row0, As, tid);

    // stage 32(d) x 16(f) slice of R[head] transposed -> Bs[wave][f][d]
#pragma unroll
    for (int it = 0; it < 4; ++it) {
        const int chunk = lane + it * 32;       // 0..127
        const int d  = chunk >> 2;              // 0..31
        const int c4 = (chunk & 3) * 4;         // 0,4,8,12
        float4 rv = *(const float4*)&R[(size_t)head * DHH * DHH + d * DHH + fc + c4];
        Bs[wave][c4 + 0][d] = (_Float16)rv.x;
        Bs[wave][c4 + 1][d] = (_Float16)rv.y;
        Bs[wave][c4 + 2][d] = (_Float16)rv.z;
        Bs[wave][c4 + 3][d] = (_Float16)rv.w;
    }
    __syncthreads();

    const int half = lane >> 4;
    const int rsel = lane & 15;

    v16h a = load_Afrag(As, rsel, half, head * DHH);
    v16h b = *(const v16h*)&Bs[wave][rsel][half * 16];
    v8f acc = {};
    acc = __builtin_amdgcn_wmma_f32_16x16x32_f16(
              false, a, false, b, (short)0, acc, false, false);

    const int col = head * DHH + fc + rsel;
#pragma unroll
    for (int i = 0; i < 8; ++i)
        Y[(size_t)(row0 + i + half * 8) * DD + col] = acc[i];
}

// ---------------------------------------------------------------------------
// Elementwise helpers
// ---------------------------------------------------------------------------
__global__ void fill_f32(float* __restrict__ p, float v, size_t n)
{
    size_t i = (size_t)blockIdx.x * blockDim.x + threadIdx.x;
    if (i < n) p[i] = v;
}

__device__ __forceinline__ void atomic_max_f32(float* addr, float v)
{
    if (v >= 0.0f) atomicMax((int*)addr, __float_as_int(v));
    else           atomicMin((unsigned int*)addr, __float_as_uint(v));
}

// ---------------------------------------------------------------------------
// Edge pass 1: sc[e,h] = (KT[src,h,:] . q[dst,h,:]) * prel[h] * scale
//              + atomic running max per destination  (KT pre-transformed)
// ---------------------------------------------------------------------------
__global__ __launch_bounds__(256)
void edge_scores(const float* __restrict__ kt, const float* __restrict__ qdst,
                 const int* __restrict__ ei, const float* __restrict__ prel,
                 float scale, int E, float* __restrict__ sc,
                 float* __restrict__ m)
{
    const int e = blockIdx.x * 256 + threadIdx.x;
    if (e >= E) return;
    const int src = ei[e];
    const int dst = ei[E + e];
    const float4* a4 = (const float4*)(kt   + (size_t)src * DD);
    const float4* b4 = (const float4*)(qdst + (size_t)dst * DD);

#pragma unroll
    for (int h = 0; h < HH; ++h) {
        float s = 0.0f;
#pragma unroll
        for (int j = 0; j < 8; ++j) {
            const float4 av = a4[h * 8 + j];
            const float4 bv = b4[h * 8 + j];
            s += av.x * bv.x + av.y * bv.y + av.z * bv.z + av.w * bv.w;
        }
        const float v = s * prel[h] * scale;
        sc[(size_t)e * HH + h] = v;
        atomic_max_f32(&m[(size_t)dst * HH + h], v);
    }
}

// ---------------------------------------------------------------------------
// Edge pass 2: sc <- exp(sc - m[dst]);  z[dst] += exp
// ---------------------------------------------------------------------------
__global__ void edge_expsum(float* __restrict__ sc, const int* __restrict__ ei,
                            const float* __restrict__ m, float* __restrict__ z,
                            int E)
{
    const int i = blockIdx.x * blockDim.x + threadIdx.x;
    if (i >= E * HH) return;
    const int e = i >> 2, h = i & 3;
    const int dst = ei[E + e];
    const float v = __expf(sc[i] - m[(size_t)dst * HH + h]);
    sc[i] = v;
    atomicAdd(&z[(size_t)dst * HH + h], v);
}

// ---------------------------------------------------------------------------
// Edge pass 3: agg[dst,h,:] += (sc/z) * VT[src,h,:]   (VT pre-transformed)
// ---------------------------------------------------------------------------
__global__ __launch_bounds__(256)
void edge_scatter(const float* __restrict__ vt, const int* __restrict__ ei,
                  const float* __restrict__ sc, const float* __restrict__ z,
                  float* __restrict__ agg, int E)
{
    const int e = blockIdx.x * 256 + threadIdx.x;
    if (e >= E) return;
    const int src = ei[e];
    const int dst = ei[E + e];
    const float4* v4 = (const float4*)(vt + (size_t)src * DD);
    float* ag = agg + (size_t)dst * DD;

#pragma unroll
    for (int h = 0; h < HH; ++h) {
        const float alpha = sc[(size_t)e * HH + h] /
                            (z[(size_t)dst * HH + h] + 1e-16f);
#pragma unroll
        for (int j = 0; j < 8; ++j) {
            const float4 vv = v4[h * 8 + j];
            const int base = h * DHH + j * 4;
            atomicAdd(&ag[base + 0], alpha * vv.x);
            atomicAdd(&ag[base + 1], alpha * vv.y);
            atomicAdd(&ag[base + 2], alpha * vv.z);
            atomicAdd(&ag[base + 3], alpha * vv.w);
        }
    }
}

// ---------------------------------------------------------------------------
// Host orchestration
// ---------------------------------------------------------------------------
static inline void run_fill(float* p, float v, size_t n, hipStream_t s)
{
    fill_f32<<<(unsigned)((n + 255) / 256), 256, 0, s>>>(p, v, n);
}

extern "C" void kernel_launch(void* const* d_in, const int* in_sizes, int n_in,
                              void* d_out, int out_size, void* d_ws, size_t ws_size,
                              hipStream_t stream)
{
    (void)in_sizes; (void)n_in; (void)out_size; (void)ws_size;

    const float* x_author = (const float*)d_in[0];
    const float* x_paper  = (const float*)d_in[1];
    const int*   ei_w     = (const int*)d_in[2];
    const int*   ei_r     = (const int*)d_in[3];
    const int*   ei_c     = (const int*)d_in[4];
    const float* proj_w   = (const float*)d_in[5];   // [2,D,D]
    const float* proj_b   = (const float*)d_in[6];   // [2,D]
    const float* kqv_w    = (const float*)d_in[7];   // [L,3,2,D,D]
    const float* kqv_b    = (const float*)d_in[8];   // [L,3,2,D]
    const float* out_w    = (const float*)d_in[9];   // [L,2,D,D]
    const float* out_b    = (const float*)d_in[10];  // [L,2,D]
    const float* rel_a    = (const float*)d_in[11];  // [L,R,H,DH,DH]
    const float* rel_m    = (const float*)d_in[12];  // [L,R,H,DH,DH]
    const float* p_rel    = (const float*)d_in[13];  // [L,R,H]
    const float* skip     = (const float*)d_in[14];  // [L,2]
    const float* cls_w    = (const float*)d_in[15];  // [D,OUT]
    const float* cls_b    = (const float*)d_in[16];  // [OUT]

    float* ws = (float*)d_ws;
    size_t off = 0;
    auto alloc = [&](size_t n) { float* p = ws + off; off += n; return p; };

    const size_t NF = (size_t)NA_ * DD;
    float* h_a   = alloc(NF);
    float* h_p   = alloc(NF);
    float* k_a   = alloc(NF);
    float* q_a   = alloc(NF);
    float* v_a   = alloc(NF);
    float* k_p   = alloc(NF);
    float* q_p   = alloc(NF);
    float* v_p   = alloc(NF);
    float* agg_a = alloc(NF);
    float* agg_p = alloc(NF);
    float* kt_w  = alloc(NF);   // k_a @ rel_a[0]
    float* kt_c  = alloc(NF);   // k_p @ rel_a[2]
    float* kt_r  = alloc(NF);   // k_p @ rel_a[1]
    float* vt_w  = alloc(NF);   // v_a @ rel_m[0]
    float* vt_c  = alloc(NF);   // v_p @ rel_m[2]
    float* vt_r  = alloc(NF);   // v_p @ rel_m[1]
    float* sc_w  = alloc((size_t)EE * HH);
    float* sc_c  = alloc((size_t)EE * HH);
    float* sc_r  = alloc((size_t)EE * HH);
    float* m_p   = alloc((size_t)NP__ * HH);
    float* z_p   = alloc((size_t)NP__ * HH);
    float* m_a   = alloc((size_t)NA_ * HH);
    float* z_a   = alloc((size_t)NA_ * HH);

    const float scale = 0.17677669529663689f;  // 1/sqrt(32)
    const float NEG_INF = -__builtin_huge_valf();

    const dim3 ggrid(NA_ / 16);   // 6250 blocks, M is an exact multiple of 16
    const int egrid  = (EE + 255) / 256;
    const int ehgrid = (EE * HH + 255) / 256;

    // ---- input projection ----
    gemm128<DD, false, false><<<ggrid, 256, 0, stream>>>(
        x_author, proj_w, proj_b, h_a, nullptr);
    gemm128<DD, false, false><<<ggrid, 256, 0, stream>>>(
        x_paper, proj_w + DD * DD, proj_b + DD, h_p, nullptr);

    for (int l = 0; l < LL; ++l) {
        // fused K/Q/V per node type
        kqv_wmma<<<ggrid, 256, 0, stream>>>(
            h_a, kqv_w + (size_t)(l * 6 + 0) * DD * DD,
            kqv_b + (size_t)(l * 6 + 0) * DD, k_a, q_a, v_a);
        kqv_wmma<<<ggrid, 256, 0, stream>>>(
            h_p, kqv_w + (size_t)(l * 6 + 1) * DD * DD,
            kqv_b + (size_t)(l * 6 + 1) * DD, k_p, q_p, v_p);

        auto ra = [&](int r) { return rel_a + ((size_t)l * 3 + r) * HH * DHH * DHH; };
        auto rm = [&](int r) { return rel_m + ((size_t)l * 3 + r) * HH * DHH * DHH; };
        auto pr = [&](int r) { return p_rel + ((size_t)l * 3 + r) * HH; };

        // hoisted per-node head transforms (WMMA, K=32 per head)
        headproj_wmma<<<ggrid, 256, 0, stream>>>(k_a, ra(0), kt_w);
        headproj_wmma<<<ggrid, 256, 0, stream>>>(k_p, ra(2), kt_c);
        headproj_wmma<<<ggrid, 256, 0, stream>>>(k_p, ra(1), kt_r);
        headproj_wmma<<<ggrid, 256, 0, stream>>>(v_a, rm(0), vt_w);
        headproj_wmma<<<ggrid, 256, 0, stream>>>(v_p, rm(2), vt_c);
        headproj_wmma<<<ggrid, 256, 0, stream>>>(v_p, rm(1), vt_r);

        run_fill(m_p, NEG_INF, (size_t)NP__ * HH, stream);
        run_fill(m_a, NEG_INF, (size_t)NA_ * HH, stream);
        run_fill(z_p, 0.0f, (size_t)NP__ * HH, stream);
        run_fill(z_a, 0.0f, (size_t)NA_ * HH, stream);
        run_fill(agg_p, 0.0f, NF, stream);
        run_fill(agg_a, 0.0f, NF, stream);

        edge_scores<<<egrid, 256, 0, stream>>>(kt_w, q_p, ei_w, pr(0), scale, EE, sc_w, m_p);
        edge_scores<<<egrid, 256, 0, stream>>>(kt_c, q_p, ei_c, pr(2), scale, EE, sc_c, m_p);
        edge_scores<<<egrid, 256, 0, stream>>>(kt_r, q_a, ei_r, pr(1), scale, EE, sc_r, m_a);

        edge_expsum<<<ehgrid, 256, 0, stream>>>(sc_w, ei_w, m_p, z_p, EE);
        edge_expsum<<<ehgrid, 256, 0, stream>>>(sc_c, ei_c, m_p, z_p, EE);
        edge_expsum<<<ehgrid, 256, 0, stream>>>(sc_r, ei_r, m_a, z_a, EE);

        edge_scatter<<<egrid, 256, 0, stream>>>(vt_w, ei_w, sc_w, z_p, agg_p, EE);
        edge_scatter<<<egrid, 256, 0, stream>>>(vt_c, ei_c, sc_c, z_p, agg_p, EE);
        edge_scatter<<<egrid, 256, 0, stream>>>(vt_r, ei_r, sc_r, z_a, agg_a, EE);

        // output mix fused: h = elu(s*(gelu(agg)@W+b) + (1-s)*h), in place on h
        gemm128<DD, true, true><<<ggrid, 256, 0, stream>>>(
            agg_a, out_w + (size_t)(l * 2 + 0) * DD * DD,
            out_b + (size_t)(l * 2 + 0) * DD, h_a, skip + l * 2 + 0);
        gemm128<DD, true, true><<<ggrid, 256, 0, stream>>>(
            agg_p, out_w + (size_t)(l * 2 + 1) * DD * DD,
            out_b + (size_t)(l * 2 + 1) * DD, h_p, skip + l * 2 + 1);
    }

    // ---- classifier: d_out = h_p @ cls_w + cls_b ----
    gemm128<OUTD, false, false><<<ggrid, 256, 0, stream>>>(
        h_p, cls_w, cls_b, (float*)d_out, nullptr);
}
